// SimpleALIGNN_75110388072869
// MI455X (gfx1250) — matrix operand order repaired and checked by reading, not compiled, not run on hardware
//
#include <hip/hip_runtime.h>
#include <hip/hip_bf16.h>
#include <stdint.h>

typedef __attribute__((ext_vector_type(16))) __bf16 v16bf;
typedef __attribute__((ext_vector_type(8)))  float  v8f;

#define H 128
#define WAVES 4
typedef unsigned short u16;

__device__ __forceinline__ unsigned f2bf1(float f){
  unsigned u = __float_as_uint(f);
  unsigned r = (u + 0x7FFFu + ((u >> 16) & 1u)) >> 16;
  return r & 0xFFFFu;
}
__device__ __forceinline__ float bf2f(unsigned h){
  return __uint_as_float((h & 0xFFFFu) << 16);
}
__device__ __forceinline__ float siluf(float x){ return x / (1.f + __expf(-x)); }
__device__ __forceinline__ float sigm(float x){ return 1.f / (1.f + __expf(-x)); }
__device__ __forceinline__ void lds_fence(){ asm volatile("s_wait_dscnt 0" ::: "memory"); }
__device__ __forceinline__ void atomAddF(float* p, float v){
  __hip_atomic_fetch_add(p, v, __ATOMIC_RELAXED, __HIP_MEMORY_SCOPE_AGENT);
}

struct Frag { union { v16bf v; unsigned u[8]; }; };

__device__ __forceinline__ v8f vzero(){
  v8f z = {0.f,0.f,0.f,0.f,0.f,0.f,0.f,0.f};
  return z;
}
__device__ __forceinline__ v8f wmma_bf(const Frag& a, const Frag& b, v8f c){
  return __builtin_amdgcn_wmma_f32_16x16x32_bf16(false, a.v, false, b.v, (short)0, c, false, false);
}
// B fragment load from pre-packed weights: 256 dwords per 32x16 frag
__device__ __forceinline__ void loadB(Frag& b, const unsigned* __restrict__ pack,
                                      int numKt, int nt, int kt, int lane){
  const unsigned* p = pack + (((size_t)(nt*numKt + kt)) << 8) + (lane << 3);
  #pragma unroll
  for (int i = 0; i < 8; ++i) b.u[i] = p[i];
}
// A fragment: row-major bf16 row pointer, 32 K-values starting at row
__device__ __forceinline__ void loadA16(Frag& a, const u16* row, int half){
  #pragma unroll
  for (int i = 0; i < 8; ++i){
    int k = ((i >> 2) << 4) + (half << 3) + ((i & 3) << 1);
    a.u[i] = *(const unsigned*)(row + k);
  }
}
__device__ __forceinline__ void loadA32(Frag& a, const float* row, int half){
  #pragma unroll
  for (int i = 0; i < 8; ++i){
    int k = ((i >> 2) << 4) + (half << 3) + ((i & 3) << 1);
    a.u[i] = f2bf1(row[k]) | (f2bf1(row[k + 1]) << 16);
  }
}

// ---------------- message + gated scatter-add kernel -----------------
// x_new[d] += sigmoid(silu(comb@g0)@g1) * (silu(silu(comb@n0)@n1)@n2)
// comb = [x[d] | x[s] | ea], one 16-edge tile per wave.
template<bool EF32>
__global__ __launch_bounds__(128)
void msg_kernel(const u16* __restrict__ nodeF,
                const u16* __restrict__ ef16, const float* __restrict__ ef32,
                const int* __restrict__ srcI, const int* __restrict__ dstI, int nE,
                const unsigned* __restrict__ n0p, const float* __restrict__ n0b,
                const unsigned* __restrict__ n1p, const float* __restrict__ n1b,
                const unsigned* __restrict__ n2p, const float* __restrict__ n2b,
                const unsigned* __restrict__ g0p, const float* __restrict__ g0b,
                const float* __restrict__ g1w, const float* __restrict__ g1b,
                float* __restrict__ acc)
{
  __shared__ u16  stage[WAVES][16 * H];
  __shared__ float gateL[WAVES][16];
  __shared__ int   dstL[WAVES][16];
  const int lane = threadIdx.x & 31;
  const int wv   = threadIdx.x >> 5;
  const int half = lane >> 4;
  const int ml   = lane & 15;
  const int tile = (blockIdx.x * WAVES + wv) * 16;
  int eRow = tile + ml; if (eRow > nE - 1) eRow = nE - 1;
  const int sI = srcI[eRow];
  const int dI = dstI[eRow];
  if (lane < 16) dstL[wv][lane] = dI;

  v8f aN[8], aG[8];
  #pragma unroll
  for (int t = 0; t < 8; ++t){ aN[t] = vzero(); aG[t] = vzero(); }

  for (int kt = 0; kt < 12; ++kt){
    Frag a;
    if (kt < 8){
      const u16* row = nodeF + (size_t)(kt < 4 ? dI : sI) * H + (kt & 3) * 32;
      loadA16(a, row, half);
    } else if (!EF32){
      loadA16(a, ef16 + (size_t)eRow * H + (kt - 8) * 32, half);
    } else {
      loadA32(a, ef32 + (size_t)eRow * H + (kt - 8) * 32, half);
    }
    #pragma unroll
    for (int nt = 0; nt < 8; ++nt){
      Frag b; loadB(b, n0p, 12, nt, kt, lane);
      aN[nt] = wmma_bf(a, b, aN[nt]);
      Frag g; loadB(g, g0p, 12, nt, kt, lane);
      aG[nt] = wmma_bf(a, g, aG[nt]);
    }
  }
  // gate hidden: silu -> LDS
  #pragma unroll
  for (int nt = 0; nt < 8; ++nt){
    #pragma unroll
    for (int v = 0; v < 8; ++v){
      int n = nt * 16 + ml, m = v + half * 8;
      stage[wv][m * H + n] = (u16)f2bf1(siluf(aG[nt][v] + g0b[n]));
    }
  }
  lds_fence();
  if (lane < 16){
    float s = g1b[0];
    const u16* r = &stage[wv][lane * H];
    for (int k = 0; k < H; ++k) s += bf2f(r[k]) * g1w[k];
    gateL[wv][lane] = sigm(s);
  }
  lds_fence();
  // h0 = silu(comb@n0+b) -> LDS
  #pragma unroll
  for (int nt = 0; nt < 8; ++nt){
    #pragma unroll
    for (int v = 0; v < 8; ++v){
      int n = nt * 16 + ml, m = v + half * 8;
      stage[wv][m * H + n] = (u16)f2bf1(siluf(aN[nt][v] + n0b[n]));
    }
  }
  lds_fence();
  // h1 = silu(h0@n1+b)
  v8f a1[8];
  #pragma unroll
  for (int t = 0; t < 8; ++t) a1[t] = vzero();
  for (int kt = 0; kt < 4; ++kt){
    Frag a; loadA16(a, &stage[wv][ml * H + kt * 32], half);
    #pragma unroll
    for (int nt = 0; nt < 8; ++nt){
      Frag b; loadB(b, n1p, 4, nt, kt, lane);
      a1[nt] = wmma_bf(a, b, a1[nt]);
    }
  }
  lds_fence();
  #pragma unroll
  for (int nt = 0; nt < 8; ++nt){
    #pragma unroll
    for (int v = 0; v < 8; ++v){
      int n = nt * 16 + ml, m = v + half * 8;
      stage[wv][m * H + n] = (u16)f2bf1(siluf(a1[nt][v] + n1b[n]));
    }
  }
  lds_fence();
  // msg = h1@n2 + b
  v8f a2[8];
  #pragma unroll
  for (int t = 0; t < 8; ++t) a2[t] = vzero();
  for (int kt = 0; kt < 4; ++kt){
    Frag a; loadA16(a, &stage[wv][ml * H + kt * 32], half);
    #pragma unroll
    for (int nt = 0; nt < 8; ++nt){
      Frag b; loadB(b, n2p, 4, nt, kt, lane);
      a2[nt] = wmma_bf(a, b, a2[nt]);
    }
  }
  // scatter-add gated messages (segment_sum at dst)
  #pragma unroll
  for (int nt = 0; nt < 8; ++nt){
    #pragma unroll
    for (int v = 0; v < 8; ++v){
      int n = nt * 16 + ml, m = v + half * 8;
      if (tile + m < nE){
        float val = (a2[nt][v] + n2b[n]) * gateL[wv][m];
        atomAddF(acc + (size_t)dstL[wv][m] * H + n, val);
      }
    }
  }
}

// ---------------- edge update: ea_new = ea + silu(comb2@e0+b)@e1 + b ----------
__global__ __launch_bounds__(128)
void edge_update_kernel(const u16* __restrict__ nodeNew,
                        const u16* __restrict__ ea16,
                        const int* __restrict__ srcI, const int* __restrict__ dstI, int nE,
                        const unsigned* __restrict__ e0p, const float* __restrict__ e0b,
                        const unsigned* __restrict__ e1p, const float* __restrict__ e1b,
                        u16* __restrict__ out)
{
  __shared__ u16 stage[WAVES][16 * H];
  const int lane = threadIdx.x & 31;
  const int wv   = threadIdx.x >> 5;
  const int half = lane >> 4;
  const int ml   = lane & 15;
  const int tile = (blockIdx.x * WAVES + wv) * 16;
  int eRow = tile + ml; if (eRow > nE - 1) eRow = nE - 1;
  const int sI = srcI[eRow];
  const int dI = dstI[eRow];

  v8f aH[8];
  #pragma unroll
  for (int t = 0; t < 8; ++t) aH[t] = vzero();
  for (int kt = 0; kt < 12; ++kt){
    Frag a;
    if (kt < 8){ // comb2 = [x_new[s] | x_new[d] | ea]
      loadA16(a, nodeNew + (size_t)(kt < 4 ? sI : dI) * H + (kt & 3) * 32, half);
    } else {
      loadA16(a, ea16 + (size_t)eRow * H + (kt - 8) * 32, half);
    }
    #pragma unroll
    for (int nt = 0; nt < 8; ++nt){
      Frag b; loadB(b, e0p, 12, nt, kt, lane);
      aH[nt] = wmma_bf(a, b, aH[nt]);
    }
  }
  #pragma unroll
  for (int nt = 0; nt < 8; ++nt){
    #pragma unroll
    for (int v = 0; v < 8; ++v){
      int n = nt * 16 + ml, m = v + half * 8;
      stage[wv][m * H + n] = (u16)f2bf1(siluf(aH[nt][v] + e0b[n]));
    }
  }
  lds_fence();
  v8f aO[8];
  #pragma unroll
  for (int t = 0; t < 8; ++t) aO[t] = vzero();
  for (int kt = 0; kt < 4; ++kt){
    Frag a; loadA16(a, &stage[wv][ml * H + kt * 32], half);
    #pragma unroll
    for (int nt = 0; nt < 8; ++nt){
      Frag b; loadB(b, e1p, 4, nt, kt, lane);
      aO[nt] = wmma_bf(a, b, aO[nt]);
    }
  }
  lds_fence();
  // restage result (+bias) as bf16 rows, then coalesced row writes with ea add
  #pragma unroll
  for (int nt = 0; nt < 8; ++nt){
    #pragma unroll
    for (int v = 0; v < 8; ++v){
      int n = nt * 16 + ml, m = v + half * 8;
      stage[wv][m * H + n] = (u16)f2bf1(aO[nt][v] + e1b[n]);
    }
  }
  lds_fence();
  int e = tile + ml;
  if (e < nE){
    const unsigned* src   = (const unsigned*)&stage[wv][ml * H] + half * 32;
    const unsigned* eaRow = (const unsigned*)(ea16 + (size_t)e * H) + half * 32;
    unsigned* dstp        = (unsigned*)(out + (size_t)e * H) + half * 32;
    #pragma unroll
    for (int i = 0; i < 32; ++i){
      unsigned aw = src[i], bw = eaRow[i];
      float lo = bf2f(aw) + bf2f(bw);
      float hi = bf2f(aw >> 16) + bf2f(bw >> 16);
      dstp[i] = f2bf1(lo) | (f2bf1(hi) << 16);
    }
  }
}

// ---------------- small utility kernels ------------------------------
__global__ void pack_kernel(const float* __restrict__ W, unsigned* __restrict__ out,
                            int K, int N){
  int t = blockIdx.x * blockDim.x + threadIdx.x;
  int numKt = K / 32;
  int total = numKt * (N / 16) * 256;
  if (t >= total) return;
  int f = t >> 8, r = t & 255, lane = r >> 3, i = r & 7;
  int kt = f % numKt, nt = f / numKt;
  int k = kt * 32 + ((i >> 2) << 4) + ((lane >> 4) << 3) + ((i & 3) << 1);
  int n = nt * 16 + (lane & 15);
  unsigned lo = f2bf1(W[(size_t)k * N + n]);
  unsigned hi = f2bf1(W[(size_t)(k + 1) * N + n]);
  out[t] = lo | (hi << 16);
}
__global__ void b2f_kernel(float* __restrict__ d, const u16* __restrict__ s, size_t n){
  size_t i = (size_t)blockIdx.x * blockDim.x + threadIdx.x;
  if (i < n) d[i] = bf2f(s[i]);
}
__global__ void f2b_kernel(u16* __restrict__ d, const float* __restrict__ s, size_t n){
  size_t i = (size_t)blockIdx.x * blockDim.x + threadIdx.x;
  if (i < n) d[i] = (u16)f2bf1(s[i]);
}
__global__ void zero_kernel(float* __restrict__ p, size_t n){
  size_t i = (size_t)blockIdx.x * blockDim.x + threadIdx.x;
  if (i < n) p[i] = 0.f;
}
__global__ void embed_bf_kernel(const float* __restrict__ x, const float* __restrict__ W,
                                const float* __restrict__ b, u16* __restrict__ out,
                                int rows, int din){
  size_t i = (size_t)blockIdx.x * blockDim.x + threadIdx.x;
  if (i >= (size_t)rows * H) return;
  int rr = (int)(i / H), j = (int)(i % H);
  float s = b[j];
  for (int k = 0; k < din; ++k) s += x[(size_t)rr * din + k] * W[k * H + j];
  out[i] = (u16)f2bf1(s);
}
__global__ void embed_f32_kernel(const float* __restrict__ x, const float* __restrict__ W,
                                 const float* __restrict__ b, float* __restrict__ out,
                                 int rows, int din){
  size_t i = (size_t)blockIdx.x * blockDim.x + threadIdx.x;
  if (i >= (size_t)rows * H) return;
  int rr = (int)(i / H), j = (int)(i % H);
  float s = b[j];
  for (int k = 0; k < din; ++k) s += x[(size_t)rr * din + k] * W[k * H + j];
  out[i] = s;
}
__global__ void pool_kernel(const float* __restrict__ hacc, const int* __restrict__ batch,
                            float* __restrict__ pool, float* __restrict__ cnt, int N){
  size_t i = (size_t)blockIdx.x * blockDim.x + threadIdx.x;
  if (i >= (size_t)N * H) return;
  int n = (int)(i / H), j = (int)(i % H);
  int b = batch[n];
  atomAddF(pool + (size_t)b * H + j, hacc[i]);
  if (j == 0) atomAddF(cnt + b, 1.0f);
}
__global__ __launch_bounds__(128)
void final_kernel(const float* __restrict__ pool, const float* __restrict__ cnt,
                  const float* __restrict__ uE,
                  const float* __restrict__ w0, const float* __restrict__ b0,
                  const float* __restrict__ w1, const float* __restrict__ b1,
                  float* __restrict__ out){
  int b = blockIdx.x, j = threadIdx.x;
  __shared__ float xc[2 * H];
  __shared__ float red[H];
  float c = fmaxf(cnt[b], 1.0f);
  xc[j]     = pool[(size_t)b * H + j] / c;
  xc[H + j] = uE[(size_t)b * H + j];
  __syncthreads();
  float s = b0[j];
  for (int k = 0; k < 2 * H; ++k) s += xc[k] * w0[(size_t)k * H + j];
  red[j] = siluf(s) * w1[j];
  __syncthreads();
  for (int off = H / 2; off > 0; off >>= 1){
    if (j < off) red[j] += red[j + off];
    __syncthreads();
  }
  if (j == 0) out[b] = red[0] + b1[0];
}

// ---------------- host orchestration ---------------------------------
extern "C" void kernel_launch(void* const* d_in, const int* in_sizes, int n_in,
                              void* d_out, int out_size, void* d_ws, size_t ws_size,
                              hipStream_t stream){
  (void)in_sizes; (void)n_in; (void)out_size; (void)ws_size;
  const int N = 50000, E = 500000, L = 500000, B = 64;
  const float* x    = (const float*)d_in[0];
  const float* ea0  = (const float*)d_in[1];
  const float* lgea = (const float*)d_in[2];
  const float* u    = (const float*)d_in[3];
  const int*   ei   = (const int*)d_in[4];
  const int*   lgei = (const int*)d_in[5];
  const int*   bat  = (const int*)d_in[6];
  auto P = [&](int i){ return (const float*)d_in[i]; };
  const float* nw = P(7),  *nb = P(8);
  const float* ew = P(9),  *eb = P(10);
  const float* gw = P(11), *gb = P(12);
  struct Conv { const float *n0w,*n0b,*n1w,*n1b,*n2w,*n2b,*g0w,*g0b,*g1w,*g1b,*e0w,*e0b,*e1w,*e1b; };
  auto getConv = [&](int s){ Conv c{P(s),P(s+1),P(s+2),P(s+3),P(s+4),P(s+5),P(s+6),
                                    P(s+7),P(s+8),P(s+9),P(s+10),P(s+11),P(s+12),P(s+13)}; return c; };
  Conv A0 = getConv(13), Li = getConv(27), A1 = getConv(41); // layer1.line (55..) is dead code
  const float* o0w = P(69), *o0b = P(70), *o1w = P(71), *o1b = P(72);

  // workspace carve
  char* wp = (char*)d_ws;
  auto alloc = [&](size_t bytes)->void* {
    void* r = (void*)wp; wp += (bytes + 255) & ~(size_t)255; return r;
  };
  float* accH = (float*)alloc((size_t)N * H * 4);
  float* accE = (float*)alloc((size_t)E * H * 4);
  u16*   hb   = (u16*)  alloc((size_t)N * H * 2);
  u16*   eab  = (u16*)  alloc((size_t)E * H * 2);
  u16*   enb  = (u16*)  alloc((size_t)E * H * 2);
  float* uE   = (float*)alloc((size_t)B * H * 4);
  float* pool = (float*)alloc(((size_t)B * H + B) * 4);
  float* cnt  = pool + (size_t)B * H;
  auto allocPack = [&](int K, int Nc){ return (unsigned*)alloc((size_t)K * Nc * 2); };
  unsigned *A0n0 = allocPack(384,128), *A0g0 = allocPack(384,128);
  unsigned *A0n1 = allocPack(128,128), *A0n2 = allocPack(128,128);
  unsigned *A0e0 = allocPack(384,128), *A0e1 = allocPack(128,128);
  unsigned *Ln0  = allocPack(384,128), *Lg0  = allocPack(384,128);
  unsigned *Ln1  = allocPack(128,128), *Ln2  = allocPack(128,128);
  unsigned *B1n0 = allocPack(384,128), *B1g0 = allocPack(384,128);
  unsigned *B1n1 = allocPack(128,128), *B1n2 = allocPack(128,128);

  auto packW = [&](const float* W, unsigned* dst, int K, int Nc){
    int total = (K / 32) * (Nc / 16) * 256;
    pack_kernel<<<(total + 255) / 256, 256, 0, stream>>>(W, dst, K, Nc);
  };
  packW(A0.n0w, A0n0, 384, 128); packW(A0.g0w, A0g0, 384, 128);
  packW(A0.n1w, A0n1, 128, 128); packW(A0.n2w, A0n2, 128, 128);
  packW(A0.e0w, A0e0, 384, 128); packW(A0.e1w, A0e1, 128, 128);
  packW(Li.n0w, Ln0, 384, 128);  packW(Li.g0w, Lg0, 384, 128);
  packW(Li.n1w, Ln1, 128, 128);  packW(Li.n2w, Ln2, 128, 128);
  packW(A1.n0w, B1n0, 384, 128); packW(A1.g0w, B1g0, 384, 128);
  packW(A1.n1w, B1n1, 128, 128); packW(A1.n2w, B1n2, 128, 128);

  const size_t nH = (size_t)N * H, eH = (size_t)E * H;
  // embeds
  embed_bf_kernel<<<(unsigned)((nH + 255) / 256), 256, 0, stream>>>(x,   nw, nb, hb,  N, 9);
  embed_bf_kernel<<<(unsigned)((eH + 255) / 256), 256, 0, stream>>>(ea0, ew, eb, eab, E, 6);
  embed_f32_kernel<<<(unsigned)((B * H + 255) / 256), 256, 0, stream>>>(u, gw, gb, uE, B, 6);

  const int blocksE = ((E + 15) / 16 + WAVES - 1) / WAVES;
  const int blocksL = ((L + 15) / 16 + WAVES - 1) / WAVES;

  // layer 1 atom conv: x_new = h + segsum(gate*msg) ; then edge update
  b2f_kernel<<<(unsigned)((nH + 255) / 256), 256, 0, stream>>>(accH, hb, nH);
  msg_kernel<false><<<blocksE, 128, 0, stream>>>(hb, eab, nullptr, ei, ei + E, E,
      A0n0, A0.n0b, A0n1, A0.n1b, A0n2, A0.n2b, A0g0, A0.g0b, A0.g1w, A0.g1b, accH);
  f2b_kernel<<<(unsigned)((nH + 255) / 256), 256, 0, stream>>>(hb, accH, nH);
  edge_update_kernel<<<blocksE, 128, 0, stream>>>(hb, eab, ei, ei + E, E,
      A0e0, A0.e0b, A0e1, A0.e1b, enb);

  // layer 1 line conv (node update only; its edge output is discarded in reference)
  b2f_kernel<<<(unsigned)((eH + 255) / 256), 256, 0, stream>>>(accE, enb, eH);
  msg_kernel<true><<<blocksL, 128, 0, stream>>>(enb, nullptr, lgea, lgei, lgei + L, L,
      Ln0, Li.n0b, Ln1, Li.n1b, Ln2, Li.n2b, Lg0, Li.g0b, Li.g1w, Li.g1b, accE);
  f2b_kernel<<<(unsigned)((eH + 255) / 256), 256, 0, stream>>>(eab, accE, eH);

  // layer 2 atom conv (node update only; edge update + layer2 line are dead code)
  // accH already holds exact f32 h after layer 1 -> accumulate on top of it
  msg_kernel<false><<<blocksE, 128, 0, stream>>>(hb, eab, nullptr, ei, ei + E, E,
      B1n0, A1.n0b, B1n1, A1.n1b, B1n2, A1.n2b, B1g0, A1.g0b, A1.g1w, A1.g1b, accH);

  // global mean pool + output head
  zero_kernel<<<(unsigned)((B * H + B + 255) / 256), 256, 0, stream>>>(pool, (size_t)B * H + B);
  pool_kernel<<<(unsigned)((nH + 255) / 256), 256, 0, stream>>>(accH, bat, pool, cnt, N);
  final_kernel<<<B, 128, 0, stream>>>(pool, cnt, uE, o0w, o0b, o1w, o1b, (float*)d_out);
}
